// EncoderRNN_45251775431124
// MI455X (gfx1250) — compile-verified, matrix operand order checked by reading, not run on hardware
//
#include <hip/hip_runtime.h>
#include <stdint.h>

#define HDIM   1024
#define GDIM   3072          // 3*H
#define LNUM   2
#define BATCH  32
#define TSEQ   256
#define BT     8192          // BATCH*TSEQ

typedef __attribute__((ext_vector_type(8)))  float  v8f;
typedef __attribute__((ext_vector_type(16))) __bf16 v16bf;
typedef __attribute__((ext_vector_type(4)))  unsigned int v4u;
typedef __attribute__((ext_vector_type(8)))  int  v8i;
typedef __attribute__((ext_vector_type(4)))  int  v4i;

#if defined(__gfx1250__) && __has_builtin(__builtin_amdgcn_tensor_load_to_lds)
#define HAVE_TDM 1
#else
#define HAVE_TDM 0
#endif

struct Frag { union { uint4 u[2]; v16bf v; }; };

__device__ __forceinline__ v8f wmma_bf16(const Frag& a, const Frag& b, v8f c) {
    return __builtin_amdgcn_wmma_f32_16x16x32_bf16(
        /*neg_a*/false, a.v, /*neg_b*/false, b.v,
        /*c_mod*/(short)0, c, /*reuse_a*/false, /*reuse_b*/false);
}

__device__ __forceinline__ uint16_t f2bf(float f) {
    unsigned u = __float_as_uint(f);
    u += 0x7fffu + ((u >> 16) & 1u);      // round-to-nearest-even
    return (uint16_t)(u >> 16);
}

#if HAVE_TDM
// TDM 2D tile load: tile_dim1 rows x tile_dim0 elems (bf16, 2B), row stride
// stride0 elems, packed contiguously into LDS at lds_addr.
// D# layout per CDNA5 ISA 8.3/8.4: group0 = {count=1, lds_addr, gaddr, type=2},
// group1 = {data_size=2B, tensor_dim0/1, tile_dim0/1, tensor_dim0_stride}.
__device__ __forceinline__ void tdm_load_2d(uint32_t lds_addr, uint64_t gaddr,
                                            uint32_t tdim0, uint32_t tdim1,
                                            uint32_t tiled0, uint32_t tiled1,
                                            uint64_t stride0) {
    v4u g0 = { 1u,                                    // count=1 (valid), user mode
               lds_addr,                              // bits 63:32 lds_addr
               (uint32_t)gaddr,                       // bits 95:64 global_addr lo
               (uint32_t)((gaddr >> 32) & 0x01FFFFFFu) | (2u << 30) }; // type=2
    v8i g1 = { (int)(1u << 16),                       // data_size=1 (2 bytes)
               (int)((tdim0 & 0xFFFFu) << 16),        // tensor_dim0 lo16
               (int)((tdim0 >> 16) | ((tdim1 & 0xFFFFu) << 16)),
               (int)((tdim1 >> 16) | (tiled0 << 16)), // tile_dim0
               (int)tiled1,                           // tile_dim1 (tile_dim2=0)
               (int)(uint32_t)stride0,                // tensor_dim0_stride lo32
               (int)((uint32_t)(stride0 >> 32) & 0xFFFFu),
               0 };
    v4i z = {0, 0, 0, 0};
#if __clang_major__ >= 23
    v8i z8 = {0, 0, 0, 0, 0, 0, 0, 0};
    __builtin_amdgcn_tensor_load_to_lds(g0, g1, z, z, z8, 0);
#else
    __builtin_amdgcn_tensor_load_to_lds(g0, g1, z, z, 0);
#endif
}
#endif

// ---------------- K0: f32 -> bf16 tensor conversion (weights) ----------------
__global__ void cvt_bf16_kernel(const float* __restrict__ w,
                                uint16_t* __restrict__ o, size_t n4) {
    size_t i = (size_t)blockIdx.x * blockDim.x + threadIdx.x;
    if (i >= n4) return;
    float4 v = reinterpret_cast<const float4*>(w)[i];
    ushort4 r;
    r.x = f2bf(v.x); r.y = f2bf(v.y); r.z = f2bf(v.z); r.w = f2bf(v.w);
    reinterpret_cast<ushort4*>(o)[i] = r;
}

// ---------------- K1: embedding gather -> bf16 x [BT,H] ----------------
__global__ void gather_kernel(const int* __restrict__ ids,
                              const float* __restrict__ emb,
                              uint16_t* __restrict__ xbf) {
    size_t i = (size_t)blockIdx.x * blockDim.x + threadIdx.x;   // over BT*H/4
    if (i >= (size_t)BT * HDIM / 4) return;
    size_t bt = i / (HDIM / 4);
    int    hc = (int)(i % (HDIM / 4)) * 4;
    int    id = ids[bt];
    float4 v = *reinterpret_cast<const float4*>(emb + (size_t)id * HDIM + hc);
    ushort4 r;
    r.x = f2bf(v.x); r.y = f2bf(v.y); r.z = f2bf(v.z); r.w = f2bf(v.w);
    *reinterpret_cast<ushort4*>(xbf + bt * HDIM + hc) = r;
}

// ---------------- K2: xg = X(bf16) @ W^T(bf16) + bias, f32 out ----------------
// X: [BT, H] row-major bf16, W: [G, H] row-major bf16 (NT GEMM), XG: [BT, G] f32
// WG: 256 threads (8 waves), 128x128 tile; wave = 64(M) x 32(N) = 4x2 WMMA tiles.
// K staged 32-deep into double-buffered LDS via TDM (wave 0), pipelined 1 ahead.
#define NCHUNK (HDIM / 32)
__global__ __launch_bounds__(256)
void gemm_xw_kernel(const uint16_t* __restrict__ X,
                    const uint16_t* __restrict__ W,
                    const float* __restrict__ bias,
                    float* __restrict__ XG) {
    __shared__ uint16_t As[2][128 * 32];
    __shared__ uint16_t Bs[2][128 * 32];

    const int tid  = threadIdx.x;
    const int lane = tid & 31;
    const int wid  = tid >> 5;
    const int wm   = wid & 1;            // 0..1
    const int wn   = wid >> 1;           // 0..3
    const int m0   = blockIdx.y * 128;
    const int n0   = blockIdx.x * 128;
    const int wrow = wm * 64;
    const int wcol = wn * 32;
    const int lrow = lane & 15;
    const int lhi  = lane >> 4;

    v8f acc[4][2] = {};

#if HAVE_TDM
    const uint64_t xga = (uint64_t)(uintptr_t)X + (uint64_t)m0 * HDIM * 2;
    const uint64_t wga = (uint64_t)(uintptr_t)W + (uint64_t)n0 * HDIM * 2;
    if (wid == 0) {   // prime chunk 0 into buffer 0
        tdm_load_2d((uint32_t)(uintptr_t)&As[0][0], xga, 32, 128, 32, 128, HDIM);
        tdm_load_2d((uint32_t)(uintptr_t)&Bs[0][0], wga, 32, 128, 32, 128, HDIM);
    }
#endif

    for (int c = 0; c < NCHUNK; ++c) {
        const int cur = c & 1;
        const int kc  = c * 32;
#if HAVE_TDM
        if (wid == 0) {
            if (c + 1 < NCHUNK) {   // issue next chunk into other buffer
                tdm_load_2d((uint32_t)(uintptr_t)&As[cur ^ 1][0],
                            xga + (uint64_t)(kc + 32) * 2, 32, 128, 32, 128, HDIM);
                tdm_load_2d((uint32_t)(uintptr_t)&Bs[cur ^ 1][0],
                            wga + (uint64_t)(kc + 32) * 2, 32, 128, 32, 128, HDIM);
                __builtin_amdgcn_s_wait_tensorcnt(2);  // chunk c landed, c+1 in flight
            } else {
                __builtin_amdgcn_s_wait_tensorcnt(0);
            }
        }
        __syncthreads();
#else
        // cooperative stage: 512 uint4 per tile, 2 per thread
#pragma unroll
        for (int i = 0; i < 2; ++i) {
            int idx = tid + i * 256;       // 0..511
            int r   = idx >> 2;            // 0..127
            int cc  = (idx & 3) * 8;       // 0,8,16,24 halves
            *reinterpret_cast<uint4*>(&As[cur][r * 32 + cc]) =
                *reinterpret_cast<const uint4*>(X + (size_t)(m0 + r) * HDIM + kc + cc);
            *reinterpret_cast<uint4*>(&Bs[cur][r * 32 + cc]) =
                *reinterpret_cast<const uint4*>(W + (size_t)(n0 + r) * HDIM + kc + cc);
            if (kc + 32 < HDIM) {
                __builtin_prefetch(X + (size_t)(m0 + r) * HDIM + kc + 32 + cc, 0, 1);
                __builtin_prefetch(W + (size_t)(n0 + r) * HDIM + kc + 32 + cc, 0, 1);
            }
        }
        __syncthreads();
#endif

        Frag a[4], b[2];
#pragma unroll
        for (int i = 0; i < 4; ++i) {
            int ar = wrow + i * 16 + lrow;
            a[i].u[0] = *reinterpret_cast<const uint4*>(&As[cur][ar * 32 + lhi * 8]);
            a[i].u[1] = *reinterpret_cast<const uint4*>(&As[cur][ar * 32 + lhi * 8 + 16]);
        }
#pragma unroll
        for (int j = 0; j < 2; ++j) {
            int br = wcol + j * 16 + lrow;
            b[j].u[0] = *reinterpret_cast<const uint4*>(&Bs[cur][br * 32 + lhi * 8]);
            b[j].u[1] = *reinterpret_cast<const uint4*>(&Bs[cur][br * 32 + lhi * 8 + 16]);
        }
#pragma unroll
        for (int i = 0; i < 4; ++i)
#pragma unroll
            for (int j = 0; j < 2; ++j)
                acc[i][j] = wmma_bf16(a[i], b[j], acc[i][j]);
        __syncthreads();    // compute done before buffer is re-filled
    }

    // epilogue: C layout M = vr + 8*lhi, N = lane&15
#pragma unroll
    for (int j = 0; j < 2; ++j) {
        int   col = n0 + wcol + j * 16 + lrow;
        float bj  = bias[col];
#pragma unroll
        for (int i = 0; i < 4; ++i) {
#pragma unroll
            for (int vr = 0; vr < 8; ++vr) {
                int row = m0 + wrow + i * 16 + vr + 8 * lhi;
                XG[(size_t)row * GDIM + col] = acc[i][j][vr] + bj;
            }
        }
    }
}

// ---------------- K3: persistent GRU recurrence (one layer) ----------------
// Grid: 32 WGs x 128 threads = 128 waves; wave = (mt in 0..1, nh in 0..63):
// 16 batch rows x 16 h-columns, computes r/z/n gate tiles, one grid barrier/step.
#define RNWG 32
__global__ __launch_bounds__(128)
void gru_recur_kernel(const uint16_t* __restrict__ Whh,   // [G, H] bf16
                      const float* __restrict__ bhh,      // [G]
                      const float* __restrict__ XG,       // [BT, G]
                      float* __restrict__ hfp,            // [B, H] f32 state
                      uint16_t* __restrict__ hbf0,        // [B, H] bf16 (even t src)
                      uint16_t* __restrict__ hbf1,        // [B, H] bf16 (odd  t src)
                      uint16_t* __restrict__ ybf,         // [BT, H] bf16 or null
                      float* __restrict__ yout,           // [BT, H] f32 or null
                      float* __restrict__ hidden_out,     // [B, H] f32
                      unsigned* __restrict__ counter) {
    __shared__ uint16_t sh[BATCH * HDIM];                 // 64 KB staged h

    const int tid  = threadIdx.x;
    const int lane = tid & 31;
    const int wl   = tid >> 5;
    const int task = blockIdx.x * 4 + wl;   // 0..127
    const int mt   = task & 1;
    const int nh   = task >> 1;             // 0..63
    const int b0   = mt * 16;
    const int lrow = lane & 15;
    const int lhi  = lane >> 4;
    const int hcol = nh * 16 + lrow;

    const uint16_t* wr = Whh + (size_t)(nh * 16 + lrow) * HDIM;
    const uint16_t* wz = wr + (size_t)1024 * HDIM;
    const uint16_t* wn = wr + (size_t)2048 * HDIM;
    const float bR = bhh[hcol];
    const float bZ = bhh[hcol + 1024];
    const float bN = bhh[hcol + 2048];

    unsigned target = 0;
    for (int t = 0; t < TSEQ; ++t) {
        const uint16_t* hsrc = (t & 1) ? hbf1 : hbf0;
        uint16_t*       hdst = (t & 1) ? hbf0 : hbf1;

        // stage h[B,H] bf16 into LDS
#if HAVE_TDM
        if (wl == 0) {    // single 1-D TDM transfer: 32768 bf16 = 64 KB
            tdm_load_2d((uint32_t)(uintptr_t)sh, (uint64_t)(uintptr_t)hsrc,
                        BATCH * HDIM, 1, BATCH * HDIM, 1, BATCH * HDIM);
            __builtin_amdgcn_s_wait_tensorcnt(0);
        }
        __syncthreads();
#else
        for (int i = tid; i < BATCH * HDIM / 8; i += 128)
            reinterpret_cast<uint4*>(sh)[i] =
                reinterpret_cast<const uint4*>(hsrc)[i];
        __syncthreads();
#endif

        v8f accR = {}, accZ = {}, accN = {};
        const int arow = b0 + lrow;
        for (int kc = 0; kc < HDIM; kc += 32) {
            Frag a, fr, fz, fn;
            a.u[0]  = *reinterpret_cast<const uint4*>(&sh[arow * HDIM + kc + lhi * 8]);
            a.u[1]  = *reinterpret_cast<const uint4*>(&sh[arow * HDIM + kc + lhi * 8 + 16]);
            fr.u[0] = *reinterpret_cast<const uint4*>(wr + kc + lhi * 8);
            fr.u[1] = *reinterpret_cast<const uint4*>(wr + kc + lhi * 8 + 16);
            fz.u[0] = *reinterpret_cast<const uint4*>(wz + kc + lhi * 8);
            fz.u[1] = *reinterpret_cast<const uint4*>(wz + kc + lhi * 8 + 16);
            fn.u[0] = *reinterpret_cast<const uint4*>(wn + kc + lhi * 8);
            fn.u[1] = *reinterpret_cast<const uint4*>(wn + kc + lhi * 8 + 16);
            accR = wmma_bf16(a, fr, accR);
            accZ = wmma_bf16(a, fz, accZ);
            accN = wmma_bf16(a, fn, accN);
        }
        __syncthreads();   // LDS reuse next iter

        // gate math + state update (each (b,hcol) owned by exactly one thread)
#pragma unroll
        for (int vr = 0; vr < 8; ++vr) {
            int    b     = b0 + vr + 8 * lhi;
            size_t btrow = (size_t)b * TSEQ + t;
            float xr = XG[btrow * GDIM + hcol];
            float xz = XG[btrow * GDIM + hcol + 1024];
            float xn = XG[btrow * GDIM + hcol + 2048];
            float r  = 1.0f / (1.0f + __expf(-(xr + accR[vr] + bR)));
            float z  = 1.0f / (1.0f + __expf(-(xz + accZ[vr] + bZ)));
            float nn = tanhf(xn + r * (accN[vr] + bN));
            float hp = hfp[(size_t)b * HDIM + hcol];
            float hn = (1.0f - z) * nn + z * hp;
            hfp[(size_t)b * HDIM + hcol]  = hn;
            hdst[(size_t)b * HDIM + hcol] = f2bf(hn);
            if (ybf)  ybf[btrow * HDIM + hcol]  = f2bf(hn);
            if (yout) yout[btrow * HDIM + hcol] = hn;
        }

        // device-wide barrier (monotonic counter, target = RNWG*(t+1))
        __threadfence();
        __syncthreads();
        target += RNWG;
        if (tid == 0) {
            atomicAdd(counter, 1u);
            while (__hip_atomic_load(counter, __ATOMIC_ACQUIRE,
                                     __HIP_MEMORY_SCOPE_AGENT) < target)
                __builtin_amdgcn_s_sleep(2);
        }
        __syncthreads();
    }

    // final hidden state (values last written by this same thread)
#pragma unroll
    for (int vr = 0; vr < 8; ++vr) {
        int b = b0 + vr + 8 * lhi;
        hidden_out[(size_t)b * HDIM + hcol] = hfp[(size_t)b * HDIM + hcol];
    }
}

// ---------------- host-side orchestration ----------------
extern "C" void kernel_launch(void* const* d_in, const int* in_sizes, int n_in,
                              void* d_out, int out_size, void* d_ws, size_t ws_size,
                              hipStream_t stream) {
    const int*   ids = (const int*)  d_in[0];
    const float* emb = (const float*)d_in[1];
    const float* Wih = (const float*)d_in[2];   // [L, 3H, H]
    const float* Whh = (const float*)d_in[3];   // [L, 3H, H]
    const float* bih = (const float*)d_in[4];   // [L, 3H]
    const float* bhh = (const float*)d_in[5];   // [L, 3H]
    float*       out = (float*)d_out;

    const size_t LGH = (size_t)LNUM * GDIM * HDIM;   // 6,291,456

    uint16_t* wih_bf = (uint16_t*)d_ws;
    uint16_t* whh_bf = wih_bf + LGH;
    uint16_t* xbuf   = whh_bf + LGH;                 // [BT,H] bf16
    uint16_t* ybuf   = xbuf + (size_t)BT * HDIM;     // [BT,H] bf16
    float*    xg     = (float*)(ybuf + (size_t)BT * HDIM);   // [BT,G] f32
    float*    hfp    = xg + (size_t)BT * GDIM;       // [B,H] f32
    uint16_t* hbf0   = (uint16_t*)(hfp + (size_t)BATCH * HDIM);
    uint16_t* hbf1   = hbf0 + (size_t)BATCH * HDIM;
    unsigned* ctrs   = (unsigned*)(hbf1 + (size_t)BATCH * HDIM);

    hipMemsetAsync(ctrs, 0, 256, stream);            // barrier counters

    // weights -> bf16
    {
        size_t n4 = LGH / 4;
        int blocks = (int)((n4 + 255) / 256);
        cvt_bf16_kernel<<<blocks, 256, 0, stream>>>(Wih, wih_bf, n4);
        cvt_bf16_kernel<<<blocks, 256, 0, stream>>>(Whh, whh_bf, n4);
    }
    // embedding gather
    {
        size_t n4 = (size_t)BT * HDIM / 4;
        gather_kernel<<<(int)((n4 + 255) / 256), 256, 0, stream>>>(ids, emb, xbuf);
    }

    float* out_x = out;                               // [B,T,H]
    float* out_h = out + (size_t)BT * HDIM;           // [L,B,H]

    const uint16_t* xin = xbuf;
    for (int l = 0; l < LNUM; ++l) {
        gemm_xw_kernel<<<dim3(GDIM / 128, BT / 128), 256, 0, stream>>>(
            xin, wih_bf + (size_t)l * GDIM * HDIM, bih + (size_t)l * GDIM, xg);

        // zero h state (hfp f32 + hbf0 bf16, contiguous)
        hipMemsetAsync(hfp, 0, (size_t)BATCH * HDIM * 4 + (size_t)BATCH * HDIM * 2,
                       stream);

        bool last = (l == LNUM - 1);
        gru_recur_kernel<<<RNWG, 128, 0, stream>>>(
            whh_bf + (size_t)l * GDIM * HDIM, bhh + (size_t)l * GDIM, xg,
            hfp, hbf0, hbf1,
            last ? nullptr : ybuf,
            last ? out_x : nullptr,
            out_h + (size_t)l * BATCH * HDIM,
            ctrs + l);
        xin = ybuf;
    }
}
